// LateralCNN_86526411145684
// MI455X (gfx1250) — matrix operand order repaired and checked by reading, not compiled, not run on hardware
//
#include <hip/hip_runtime.h>
#include <hip/hip_bf16.h>

typedef __attribute__((ext_vector_type(16))) _Float16 v16h;
typedef __attribute__((ext_vector_type(8)))  _Float16 v8h;
typedef __attribute__((ext_vector_type(4)))  _Float16 v4h;
typedef __attribute__((ext_vector_type(8)))  float    v8f;
typedef __attribute__((ext_vector_type(8)))  unsigned short v8us;
typedef __attribute__((ext_vector_type(4)))  unsigned int   v4u;
typedef __attribute__((ext_vector_type(8)))  int            v8i;
typedef __attribute__((ext_vector_type(4)))  int            v4i;

#define WMMA_F32_F16(a, b, c) \
    __builtin_amdgcn_wmma_f32_16x16x32_f16(false, (a), false, (b), (short)0, (c), false, false)

// ---------------------------------------------------------------------------
// Kernel 1: conv1 (1->10, 5x5 valid) + bias + ReLU + maxpool2
// x: [B,1,28,28] f32  ->  p1: [B,10,12,12] f16
// 2 images per block; each thread owns one pooled pixel across all 10 channels
// and caches its 6x6 input patch in registers (reused 10x).
// ---------------------------------------------------------------------------
__global__ __launch_bounds__(256) void k_conv1(const float* __restrict__ x,
                                               const float* __restrict__ w1,
                                               const float* __restrict__ b1,
                                               _Float16* __restrict__ p1,
                                               int Bn)
{
    __shared__ __align__(16) float sx[1568];   // two 28x28 images
    __shared__ float sw[250];
    __shared__ float sb[10];
    const int tid = threadIdx.x;
    const int blk = blockIdx.x;
    const size_t gb  = (size_t)blk * 1568;
    const size_t lim = (size_t)Bn * 784;
    for (int i = tid; i < 1568; i += 256) {
        size_t g = gb + i;
        sx[i] = (g < lim) ? x[g] : 0.0f;
    }
    if (tid < 250) sw[tid] = w1[tid];
    if (tid < 10)  sb[tid] = b1[tid];
    __syncthreads();

    for (int i = tid; i < 288; i += 256) {
        const int imgL = (i >= 144) ? 1 : 0;
        const int pos  = i - imgL * 144;
        const int py = pos / 12, px = pos % 12;
        const float* xi = &sx[imgL * 784];
        float patch[36];
#pragma unroll
        for (int r = 0; r < 6; ++r)
#pragma unroll
            for (int c = 0; c < 6; ++c)
                patch[r * 6 + c] = xi[(2 * py + r) * 28 + (2 * px + c)];
        const int img = blk * 2 + imgL;
        if (img < Bn) {
            for (int ch = 0; ch < 10; ++ch) {
                float m = -3.4e38f;
#pragma unroll
                for (int dy = 0; dy < 2; ++dy)
#pragma unroll
                    for (int dx = 0; dx < 2; ++dx) {
                        float acc = sb[ch];
#pragma unroll
                        for (int ky = 0; ky < 5; ++ky)
#pragma unroll
                            for (int kx = 0; kx < 5; ++kx)
                                acc = fmaf(patch[(dy + ky) * 6 + (dx + kx)],
                                           sw[ch * 25 + ky * 5 + kx], acc);
                        m = fmaxf(m, acc);
                    }
                p1[(size_t)img * 1440 + ch * 144 + pos] = (_Float16)fmaxf(m, 0.0f);
            }
        }
    }
}

// ---------------------------------------------------------------------------
// Kernel 2: conv2 (10->20, 5x5 valid) + bias + ReLU + maxpool2, via WMMA
// im2col GEMM: rows = output pixels (8x8 per image), K=250 (pad 256),
// N=20 (pad 32 -> B panel zero-padded so fragment loads are unconditional).
// Block = 2 images (8 waves); wave w -> image w/4, pool-row t = w%4.
// Activations staged to LDS via TDM tensor_load_to_lds (TENSORcnt); im2col
// k -> LDS-offset map precomputed as a u16 table.
// ---------------------------------------------------------------------------
__global__ __launch_bounds__(256) void k_conv2(const _Float16* __restrict__ p1,
                                               const float* __restrict__ w2,
                                               const float* __restrict__ b2,
                                               _Float16* __restrict__ p2,
                                               int Bn)
{
    __shared__ __align__(16) _Float16 sx[2880];        // two images [2][10][12][12]
    __shared__ __align__(16) _Float16 sw[8192];        // w2 f16, padded [32][256]
    __shared__ __align__(16) unsigned short offt[256]; // im2col k -> offset (0 if pad)
    const int tid = threadIdx.x;
    const int blk = blockIdx.x;

#if __has_builtin(__builtin_amdgcn_tensor_load_to_lds)
    // --- TDM DMA: 5760 B of activations -> LDS, issued once by wave 0 ------
    if (tid < 32) {
        const unsigned lds_addr = (unsigned)(size_t)&sx[0];
        const unsigned long long ga =
            (unsigned long long)(size_t)(p1 + (size_t)blk * 2880);
        v4u g0;
        g0[0] = 1u;                                   // count=1, user mode
        g0[1] = lds_addr;                             // lds_addr
        g0[2] = (unsigned)ga;                         // global_addr[31:0]
        g0[3] = (unsigned)((ga >> 32) & 0x1FFFFFFull) // global_addr[56:32]
              | (2u << 30);                           // type=2 (image)
        v8i g1;
        g1[0] = (int)(3u << 16);                      // data_size=8B
        g1[1] = (int)(720u << 16);                    // tensor_dim0[15:0]=720
        g1[2] = (int)(1u << 16);                      // tensor_dim1[15:0]=1
        g1[3] = (int)(720u << 16);                    // tile_dim0=720
        g1[4] = 1;                                    // tile_dim1=1, tile_dim2=0
        g1[5] = 720;                                  // tensor_dim0_stride=720
        g1[6] = 0;
        g1[7] = 0;
        v4i g2 = {0, 0, 0, 0};
        v4i g3 = {0, 0, 0, 0};
        v8i g4 = {0, 0, 0, 0, 0, 0, 0, 0};
        __builtin_amdgcn_tensor_load_to_lds(g0, g1, g2, g3, g4, 0);
    }
#else
    {
        const size_t gb = (size_t)blk * 2880;
        for (int i = tid; i < 2880; i += 256) sx[i] = p1[gb + i];
    }
#endif

    // weights -> f16 LDS, padded to [32][256] with zeros
    for (int i = tid; i < 8192; i += 256) {
        int ch = i >> 8, k = i & 255;
        sw[i] = (ch < 20 && k < 250) ? (_Float16)w2[ch * 250 + k] : (_Float16)0;
    }
    // im2col offset table (one entry per thread)
    {
        const int k = tid;
        unsigned short o = 0;
        if (k < 250) {
            const int c   = k / 25;
            const int rem = k - 25 * c;
            const int ky  = rem / 5;
            const int kx  = rem - 5 * ky;
            o = (unsigned short)(c * 144 + ky * 12 + kx);
        }
        offt[tid] = o;
    }

#if __has_builtin(__builtin_amdgcn_tensor_load_to_lds)
    if (tid < 32) __builtin_amdgcn_s_wait_tensorcnt(0);
#endif
    __syncthreads();

    const int wave = tid >> 5;
    const int lane = tid & 31;
    const int laneHi = lane >> 4;
    const int ln = lane & 15;
    const int imgL = wave >> 2;
    const int t = wave & 3;          // pool row
    const int img = blk * 2 + imgL;

    v8f c0, c1;
    {
        const int ch1 = 16 + ln;
        const float bb0 = b2[ln];
        const float bb1 = (ch1 < 20) ? b2[ch1] : 0.0f;
#pragma unroll
        for (int r = 0; r < 8; ++r) { c0[r] = bb0; c1[r] = bb1; }
    }

    const _Float16* sxi = &sx[imgL * 1440];
    const int m    = ln;                  // A row within tile
    const int y0   = 2 * t + (m >> 3);    // conv output y
    const int x0   = m & 7;               // conv output x
    const int base = y0 * 12 + x0;

#pragma unroll
    for (int s = 0; s < 8; ++s) {
        // A fragment: two contiguous 8-wide k-groups per lane
        const int kb0 = 32 * s + (laneHi ? 8 : 0);
        const int kb1 = 32 * s + (laneHi ? 24 : 16);
        const v8us o0 = *(const v8us*)&offt[kb0];
        const v8us o1 = *(const v8us*)&offt[kb1];
        v16h a;
#pragma unroll
        for (int i = 0; i < 8; ++i) {
            _Float16 v0 = sxi[base + (int)o0[i]];
            _Float16 v1 = sxi[base + (int)o1[i]];
            if (s == 7) {   // compile-time: only last step has K padding
                if (kb0 + i >= 250) v0 = (_Float16)0;
                if (kb1 + i >= 250) v1 = (_Float16)0;
            }
            a[i]     = v0;
            a[8 + i] = v1;
        }
        // B fragments: zero-padded panel -> unconditional b128 LDS loads
#pragma unroll
        for (int nt = 0; nt < 2; ++nt) {
            const int ch = nt * 16 + ln;
            const v8h* wp = (const v8h*)&sw[ch * 256 + 32 * s + (laneHi ? 8 : 0)];
            const v8h blo = wp[0];
            const v8h bhi = wp[2];
            v16h b;
#pragma unroll
            for (int i = 0; i < 8; ++i) { b[i] = blo[i]; b[8 + i] = bhi[i]; }
            if (nt == 0) c0 = WMMA_F32_F16(a, b, c0);
            else         c1 = WMMA_F32_F16(a, b, c1);
        }
    }

    // maxpool (within-lane regs + lane^16 shuffle), ReLU, packed b64 store
    const bool doStore = (img < Bn) && (laneHi == 0);
#pragma unroll
    for (int nt = 0; nt < 2; ++nt) {
        const int ch = nt * 16 + ln;
        v4h pk;
#pragma unroll
        for (int xq = 0; xq < 4; ++xq) {
            float v = fmaxf(nt ? c1[2 * xq] : c0[2 * xq],
                            nt ? c1[2 * xq + 1] : c0[2 * xq + 1]);
            const float o = __shfl_xor(v, 16, 32);
            pk[xq] = (_Float16)fmaxf(fmaxf(v, o), 0.0f);
        }
        if (doStore && ch < 20)
            *(v4h*)&p2[(size_t)img * 320 + ch * 16 + t * 4] = pk;
    }
}

// ---------------------------------------------------------------------------
// Kernel 3: fc1 (320->50) + ReLU + fc2 (50->10), both via WMMA.
// 128 rows per block (8 waves x 16-row M-tiles). Weight panels zero-padded
// ([64][320] and [16][64]) so all B-fragment loads are unconditional.
// ---------------------------------------------------------------------------
__global__ __launch_bounds__(256) void k_fc(const _Float16* __restrict__ p2,
                                            const float* __restrict__ wl1,
                                            const float* __restrict__ bl1,
                                            const float* __restrict__ wl2,
                                            const float* __restrict__ bl2,
                                            float* __restrict__ out,
                                            int Bn)
{
    __shared__ __align__(16) _Float16 sw1[20480];     // wl1 f16, padded [64][320]
    __shared__ __align__(16) _Float16 sw2[1024];      // wl2 f16, padded [16][64]
    __shared__ __align__(16) _Float16 sh[8][16][64];  // per-wave hidden rows (padded)
    const int tid = threadIdx.x;
    for (int i = tid; i < 16000; i += 256) sw1[i] = (_Float16)wl1[i];
    for (int i = 16000 + tid; i < 20480; i += 256) sw1[i] = (_Float16)0;
    for (int i = tid; i < 1024; i += 256) {
        int n = i >> 6, k = i & 63;
        sw2[i] = (n < 10 && k < 50) ? (_Float16)wl2[n * 50 + k] : (_Float16)0;
    }
    __syncthreads();

    const int wave = tid >> 5;
    const int lane = tid & 31;
    const int laneHi = lane >> 4;
    const int ln = lane & 15;
    const int rowBase = blockIdx.x * 128 + wave * 16;
    const int row = rowBase + ln;
    const bool rv = (row < Bn);

    v8f acc[4];
#pragma unroll
    for (int nt = 0; nt < 4; ++nt) {
        const int ch = nt * 16 + ln;
        const float bb = (ch < 50) ? bl1[ch] : 0.0f;
#pragma unroll
        for (int r = 0; r < 8; ++r) acc[nt][r] = bb;
    }

    const _Float16* rowp = p2 + (size_t)row * 320;
    for (int s = 0; s < 10; ++s) {
        v16h a;
        if (rv) {
            const int baseo = 32 * s + (laneHi ? 8 : 0);
            const v8h lo = *(const v8h*)(rowp + baseo);
            const v8h hi = *(const v8h*)(rowp + baseo + 16);
#pragma unroll
            for (int i = 0; i < 8; ++i) { a[i] = lo[i]; a[8 + i] = hi[i]; }
        } else {
#pragma unroll
            for (int i = 0; i < 16; ++i) a[i] = (_Float16)0;
        }
#pragma unroll
        for (int nt = 0; nt < 4; ++nt) {
            const int ch = nt * 16 + ln;
            const v8h* wp = (const v8h*)&sw1[ch * 320 + 32 * s + (laneHi ? 8 : 0)];
            const v8h blo = wp[0];
            const v8h bhi = wp[2];
            v16h b;
#pragma unroll
            for (int i = 0; i < 8; ++i) { b[i] = blo[i]; b[8 + i] = bhi[i]; }
            acc[nt] = WMMA_F32_F16(a, b, acc[nt]);
        }
    }

    // ReLU -> LDS (row-major, zero-padded to K=64 for fc2)
#pragma unroll
    for (int nt = 0; nt < 4; ++nt) {
        const int ch = nt * 16 + ln;
#pragma unroll
        for (int r = 0; r < 8; ++r) {
            const int mm = r + (laneHi ? 8 : 0);
            const float v = fmaxf(acc[nt][r], 0.0f);
            sh[wave][mm][ch] = (ch < 50) ? (_Float16)v : (_Float16)0;
        }
    }
    __syncthreads();

    // fc2: K=50 (pad 64 -> 2 steps), N=10 (one 16-wide tile)
    v8f d;
    {
        const float bb = (ln < 10) ? bl2[ln] : 0.0f;
#pragma unroll
        for (int r = 0; r < 8; ++r) d[r] = bb;
    }
#pragma unroll
    for (int s = 0; s < 2; ++s) {
        v16h a;
        {
            const _Float16* hp = &sh[wave][ln][0];
            const int baseo = 32 * s + (laneHi ? 8 : 0);
            const v8h lo = *(const v8h*)(hp + baseo);
            const v8h hi = *(const v8h*)(hp + baseo + 16);
#pragma unroll
            for (int i = 0; i < 8; ++i) { a[i] = lo[i]; a[8 + i] = hi[i]; }
        }
        v16h b;
        {
            const v8h* wp = (const v8h*)&sw2[ln * 64 + 32 * s + (laneHi ? 8 : 0)];
            const v8h blo = wp[0];
            const v8h bhi = wp[2];
#pragma unroll
            for (int i = 0; i < 8; ++i) { b[i] = blo[i]; b[8 + i] = bhi[i]; }
        }
        d = WMMA_F32_F16(a, b, d);
    }

#pragma unroll
    for (int r = 0; r < 8; ++r) {
        const int mm = r + (laneHi ? 8 : 0);
        const int rr = rowBase + mm;
        if (ln < 10 && rr < Bn)
            out[(size_t)rr * 10 + ln] = d[r];
    }
}

// ---------------------------------------------------------------------------
extern "C" void kernel_launch(void* const* d_in, const int* in_sizes, int n_in,
                              void* d_out, int out_size, void* d_ws, size_t ws_size,
                              hipStream_t stream)
{
    const float* x   = (const float*)d_in[0];
    const float* w1  = (const float*)d_in[1];
    const float* b1  = (const float*)d_in[2];
    const float* w2  = (const float*)d_in[3];
    const float* b2  = (const float*)d_in[4];
    const float* wl1 = (const float*)d_in[5];
    const float* bl1 = (const float*)d_in[6];
    const float* wl2 = (const float*)d_in[7];
    const float* bl2 = (const float*)d_in[8];
    float* out = (float*)d_out;

    const int Bn = in_sizes[0] / (28 * 28);

    _Float16* p1 = (_Float16*)d_ws;                                  // [B,10,12,12]
    _Float16* p2 = (_Float16*)((char*)d_ws + (size_t)Bn * 1440 * 2); // [B,320]

    const int g1 = (Bn + 1) / 2;
    k_conv1<<<g1, 256, 0, stream>>>(x, w1, b1, p1, Bn);
    const int g2 = (Bn + 1) / 2;
    k_conv2<<<g2, 256, 0, stream>>>(p1, w2, b2, p2, Bn);
    const int g3 = (Bn + 127) / 128;
    k_fc<<<g3, 256, 0, stream>>>(p2, wl1, bl1, wl2, bl2, out, Bn);
}
